// CaMoE_Block_70617852281186
// MI455X (gfx1250) — compile-verified, matrix operand order checked by reading, not run on hardware
//
#include <hip/hip_runtime.h>
#include <stdint.h>

#define C_DIM 1024
#define H_DIM 4096
#define N_TOK 8192
#define LDS_STRIDE 40   // padded row stride (ushorts): 80B, multiple of 16B, staggers banks

typedef __attribute__((ext_vector_type(16))) __bf16 v16bf;
typedef __attribute__((ext_vector_type(8)))  float  v8f;
typedef int b128v __attribute__((vector_size(16)));   // matches async-LDS builtin param type

#define AS1 __attribute__((address_space(1)))
#define AS3 __attribute__((address_space(3)))

#if defined(__has_builtin)
# if __has_builtin(__builtin_amdgcn_global_load_async_to_lds_b128) && __has_builtin(__builtin_amdgcn_s_wait_asynccnt)
#  define USE_ASYNC 1
# endif
#endif
#ifndef USE_ASYNC
# define USE_ASYNC 0
#endif

#if USE_ASYNC
# define NBUF 4    // 4 buffers, 3 chunks in flight; nChunks is always a multiple of 4 here
#else
# define NBUF 2
#endif

__device__ __forceinline__ unsigned short f2bf(float f) {
    union { float f; unsigned int u; } v; v.f = f;
    unsigned int u = v.u;
    return (unsigned short)((u + 0x7FFFu + ((u >> 16) & 1u)) >> 16);
}
__device__ __forceinline__ float bf2f(unsigned short b) {
    union { unsigned int u; float f; } v; v.u = ((unsigned int)b) << 16; return v.f;
}
__device__ __forceinline__ float sigmoidf_(float x) {
    return __builtin_amdgcn_rcpf(1.0f + __expf(-x));   // v_rcp_f32, not IEEE div chain
}

// ------- fp32 -> bf16 convert WITH transpose: src is Kd x Nd, dst is Nd x Kd -------
__global__ __launch_bounds__(256)
void k_cvt_t(const float* __restrict__ src, unsigned short* __restrict__ dst, int Kd, int Nd) {
    __shared__ unsigned short t[32][33];
    int kb = blockIdx.x * 32, nb = blockIdx.y * 32;
    int tx = threadIdx.x & 31, ty = threadIdx.x >> 5;   // ty: 0..7
#pragma unroll
    for (int i = 0; i < 4; i++) {
        int k = kb + ty + i * 8;
        t[tx][ty + i * 8] = f2bf(src[(size_t)k * Nd + nb + tx]);   // t[n_local][k_local]
    }
    __syncthreads();
#pragma unroll
    for (int i = 0; i < 4; i++) {
        int n = nb + ty + i * 8;
        dst[(size_t)n * Kd + kb + tx] = t[ty + i * 8][tx];
    }
}

__global__ void k_zero_counts(int* c) { if (threadIdx.x < 4) c[threadIdx.x] = 0; }

// ---------------- LayerNorm: one 256-thread block per token ----------------
__global__ void k_layernorm(const float* __restrict__ x, const float* __restrict__ w,
                            const float* __restrict__ b, unsigned short* __restrict__ outBF) {
    __shared__ float red[256];
    int t = blockIdx.x, tid = threadIdx.x;
    const float* row = x + (size_t)t * C_DIM;
    float v0[4];
    float s = 0.f;
#pragma unroll
    for (int i = 0; i < 4; i++) { v0[i] = row[tid + i * 256]; s += v0[i]; }
    red[tid] = s; __syncthreads();
    for (int k = 128; k > 0; k >>= 1) { if (tid < k) red[tid] += red[tid + k]; __syncthreads(); }
    float mean = red[0] * (1.0f / C_DIM); __syncthreads();
    float vs = 0.f;
#pragma unroll
    for (int i = 0; i < 4; i++) { float dd = v0[i] - mean; vs += dd * dd; }
    red[tid] = vs; __syncthreads();
    for (int k = 128; k > 0; k >>= 1) { if (tid < k) red[tid] += red[tid + k]; __syncthreads(); }
    float inv = rsqrtf(red[0] * (1.0f / C_DIM) + 1e-5f);
#pragma unroll
    for (int i = 0; i < 4; i++) {
        int c = tid + i * 256;
        outBF[(size_t)t * C_DIM + c] = f2bf((v0[i] - mean) * inv * w[c] + b[c]);
    }
}

// ---------------- routing: conf / difficulty / affinity / argmax / compaction ----------------
__global__ __launch_bounds__(256)
void k_route(const unsigned short* __restrict__ h,
             const float* __restrict__ conf_rwkv,   // (2, C)
             const float* __restrict__ conf_trans,  // (C,)
             const float* __restrict__ w_diff,      // (C,)
             const float* __restrict__ W_aff,       // (C, 3)
             const float* __restrict__ capital,     // (3,)
             float* __restrict__ scaleArr,
             int* __restrict__ lists,               // 3 x N_TOK
             int* __restrict__ counts) {
    __shared__ float part[8][7];
    int t = blockIdx.x, tid = threadIdx.x;
    int lane = tid & 31, wv = tid >> 5;
    const unsigned short* hr = h + (size_t)t * C_DIM;
    float a[7] = { 0, 0, 0, 0, 0, 0, 0 };
    for (int i = tid; i < C_DIM; i += 256) {
        float hv = bf2f(hr[i]);
        a[0] += hv * conf_rwkv[i];
        a[1] += hv * conf_rwkv[C_DIM + i];
        a[2] += hv * conf_trans[i];
        a[3] += hv * w_diff[i];
        a[4] += hv * W_aff[i * 3 + 0];
        a[5] += hv * W_aff[i * 3 + 1];
        a[6] += hv * W_aff[i * 3 + 2];
    }
#pragma unroll
    for (int j = 0; j < 7; j++) {
#pragma unroll
        for (int o = 16; o > 0; o >>= 1) a[j] += __shfl_xor(a[j], o, 32);
        if (lane == 0) part[wv][j] = a[j];
    }
    __syncthreads();
    if (tid == 0) {
        float tot[7];
#pragma unroll
        for (int j = 0; j < 7; j++) {
            float s = 0.f;
#pragma unroll
            for (int w2 = 0; w2 < 8; w2++) s += part[w2][j];
            tot[j] = s;
        }
        float conf[3] = { sigmoidf_(tot[0]), sigmoidf_(tot[1]), sigmoidf_(tot[2]) };
        float diff = sigmoidf_(tot[3]);
        int wbest = 0;
        float best = conf[0] * capital[0] * diff + tot[4];
        for (int e = 1; e < 3; e++) {
            float bid = conf[e] * capital[e] * diff + tot[4 + e];
            if (bid > best) { best = bid; wbest = e; }   // strict > == argmax-first-max
        }
        float wc = conf[wbest];
        scaleArr[t] = wc / (wc + 1e-6f);
        int pos = atomicAdd(&counts[wbest], 1);
        lists[wbest * N_TOK + pos] = t;
    }
}

// ---------------- WMMA GEMM: out = epilogue(A @ Bt^T) ----------------
// A:  M x K bf16, row-major (gatherable rows)
// Bt: N x K bf16 (weights pre-transposed at convert time) -> staged exactly like A
// Block: 128 thr (4 waves), tile 64(M) x 128(N), wave tile 32x64, 4-deep async LDS pipeline.
enum { EPI_BF16 = 0, EPI_SIGMOID, EPI_MULAUX, EPI_SIGMULAUX, EPI_ADDX, EPI_RELUSQ, EPI_SCATTER };

__device__ __forceinline__ v16bf load_frag(const unsigned short* tile, int subRow, int lane) {
    // 16-bit operand layout: lane l<16 -> row subRow+l, K {0..7, 16..23};
    //                        lane l>=16 -> row subRow+(l-16), K {8..15, 24..31}
    int l  = lane & 15;
    int kb = (lane < 16) ? 0 : 8;
    const unsigned short* p = tile + (subRow + l) * LDS_STRIDE + kb;
    union { v16bf v; uint4 q[2]; } u;
    u.q[0] = *(const uint4*)(p);
    u.q[1] = *(const uint4*)(p + 16);
    return u.v;
}

#define WMMA_BF16(a, b, c) __builtin_amdgcn_wmma_f32_16x16x32_bf16(false, (a), false, (b), (short)0, (c), false, false)

struct Frags { v16bf a0, a1, b0, b1, b2, b3; };

__device__ __forceinline__ Frags frag_load_all(const unsigned short* bA, const unsigned short* bB,
                                               int wrow, int wcol, int lane) {
    Frags f;
    f.a0 = load_frag(bA, wrow,      lane);
    f.a1 = load_frag(bA, wrow + 16, lane);
    f.b0 = load_frag(bB, wcol,      lane);
    f.b1 = load_frag(bB, wcol + 16, lane);
    f.b2 = load_frag(bB, wcol + 32, lane);
    f.b3 = load_frag(bB, wcol + 48, lane);
    return f;
}
__device__ __forceinline__ void wmma_all(const Frags& f, v8f (&acc)[2][4]) {
    acc[0][0] = WMMA_BF16(f.a0, f.b0, acc[0][0]);
    acc[0][1] = WMMA_BF16(f.a0, f.b1, acc[0][1]);
    acc[0][2] = WMMA_BF16(f.a0, f.b2, acc[0][2]);
    acc[0][3] = WMMA_BF16(f.a0, f.b3, acc[0][3]);
    acc[1][0] = WMMA_BF16(f.a1, f.b0, acc[1][0]);
    acc[1][1] = WMMA_BF16(f.a1, f.b1, acc[1][1]);
    acc[1][2] = WMMA_BF16(f.a1, f.b2, acc[1][2]);
    acc[1][3] = WMMA_BF16(f.a1, f.b3, acc[1][3]);
}

#if USE_ASYNC
template<int OFF>
__device__ __forceinline__ void asy16(const unsigned short* g, unsigned short* l) {
    // param types revealed by clang diagnostic: (AS1 b128v*, AS3 b128v*, imm offset, imm cpol)
    __builtin_amdgcn_global_load_async_to_lds_b128((AS1 b128v*)g, (AS3 b128v*)l, OFF, 0);
}
#endif

template<int EPI, bool GATHER>
__global__ __launch_bounds__(128)
void k_gemm(const unsigned short* __restrict__ A,
            const unsigned short* __restrict__ Bt,
            int K, int N,
            const int* __restrict__ cntPtr, int Mmax,
            const int* __restrict__ list,
            const unsigned short* __restrict__ auxBF,
            const float* __restrict__ auxF,
            const float* __restrict__ scaleArr,
            unsigned short* __restrict__ outBF,
            float* __restrict__ outF) {
    int M = cntPtr ? *cntPtr : Mmax;
    int m0 = blockIdx.x * 64;
    if (m0 >= M) return;
    int n0 = blockIdx.y * 128;

    __shared__ __align__(16) unsigned short lA[NBUF][64 * LDS_STRIDE];
    __shared__ __align__(16) unsigned short lB[NBUF][128 * LDS_STRIDE];

    int tid = threadIdx.x, lane = tid & 31, wave = tid >> 5;
    int wrow = (wave >> 1) * 32;
    int wcol = (wave & 1) * 64;

    v8f acc[2][4] = {};

    // A staging: thread -> (row 0..63, half 0/1 of 32 k's); B staging: thread -> n-row tid (32 k's).
    // Out-of-range M rows are CLAMPED (not zeroed): A row m only feeds D row m, and the
    // epilogue masks rows >= M, so garbage rows never reach memory.
    int sRow = tid >> 1, sHalf = tid & 1;
    int gi = m0 + sRow;
    int gidx = (gi < M) ? gi : (M - 1);
    int gRow = GATHER ? list[gidx] : gidx;
    const unsigned short* Abase = A + (size_t)gRow * K + sHalf * 16;
    const unsigned short* Bbase = Bt + (size_t)(n0 + tid) * K;

    unsigned short* laS[NBUF];
    unsigned short* lbS[NBUF];
#pragma unroll
    for (int b = 0; b < NBUF; b++) {
        laS[b] = &lA[b][sRow * LDS_STRIDE + sHalf * 16];
        lbS[b] = &lB[b][tid * LDS_STRIDE];
    }

    int nChunks = K >> 5;   // K is 1024 or 4096 -> nChunks is 32 or 128 (multiple of 4, >= 8)

#if USE_ASYNC
    // ---- async DMA staging, 4 buffers / 3 chunks in flight.  ASYNCcnt retires in order and
    // each chunk is 6 ops, so wait(<=12) at iter c guarantees chunk c landed while chunks
    // c+1, c+2 keep streaming.  Group-of-4 unroll keeps all buffer indices compile-time.
    {
        asy16<0>(Abase,      laS[0]); asy16<16>(Abase,      laS[0]);
        asy16<0>(Bbase,      lbS[0]); asy16<16>(Bbase,      lbS[0]);
        asy16<32>(Bbase,     lbS[0]); asy16<48>(Bbase,      lbS[0]);
        asy16<0>(Abase + 32, laS[1]); asy16<16>(Abase + 32, laS[1]);
        asy16<0>(Bbase + 32, lbS[1]); asy16<16>(Bbase + 32, lbS[1]);
        asy16<32>(Bbase + 32, lbS[1]); asy16<48>(Bbase + 32, lbS[1]);
        asy16<0>(Abase + 64, laS[2]); asy16<16>(Abase + 64, laS[2]);
        asy16<0>(Bbase + 64, lbS[2]); asy16<16>(Bbase + 64, lbS[2]);
        asy16<32>(Bbase + 64, lbS[2]); asy16<48>(Bbase + 64, lbS[2]);
    }
    int cb = 0;
    for (; cb + 7 <= nChunks; cb += 4) {        // all groups whose staging targets exist
#pragma unroll
        for (int u = 0; u < 4; u++) {
            int c = cb + u;                      // reads buffer u (cb is a multiple of 4)
            __builtin_amdgcn_s_wait_asynccnt(12);
            __syncthreads();
            int kb = (c + 3) << 5;
            unsigned short* da = laS[(u + 3) & 3];
            unsigned short* db = lbS[(u + 3) & 3];
            asy16<0>(Abase + kb, da);  asy16<16>(Abase + kb, da);
            asy16<0>(Bbase + kb, db);  asy16<16>(Bbase + kb, db);
            asy16<32>(Bbase + kb, db); asy16<48>(Bbase + kb, db);
            Frags f = frag_load_all(lA[u], lB[u], wrow, wcol, lane);
            wmma_all(f, acc);
        }
    }
    // tail group: cb == nChunks - 4; tapered waits, single remaining stage (chunk nChunks-1)
    {
#pragma unroll
        for (int u = 0; u < 4; u++) {
            if (u <= 1)      __builtin_amdgcn_s_wait_asynccnt(12);
            else if (u == 2) __builtin_amdgcn_s_wait_asynccnt(6);
            else             __builtin_amdgcn_s_wait_asynccnt(0);
            __syncthreads();
            if (u == 0) {
                int kb = (cb + 3) << 5;
                unsigned short* da = laS[3];
                unsigned short* db = lbS[3];
                asy16<0>(Abase + kb, da);  asy16<16>(Abase + kb, da);
                asy16<0>(Bbase + kb, db);  asy16<16>(Bbase + kb, db);
                asy16<32>(Bbase + kb, db); asy16<48>(Bbase + kb, db);
            }
            Frags f = frag_load_all(lA[u], lB[u], wrow, wcol, lane);
            wmma_all(f, acc);
        }
    }
#else
    // ---- register staging path (double buffered, last iteration peeled) ----
    {
        const uint4* pa = (const uint4*)(Abase);
        uint4 a0 = pa[0], a1 = pa[1];
        const uint4* pb = (const uint4*)(Bbase);
        uint4 b0 = pb[0], b1 = pb[1], b2 = pb[2], b3 = pb[3];
        uint4* dA = (uint4*)laS[0]; dA[0] = a0; dA[1] = a1;
        uint4* dB = (uint4*)lbS[0]; dB[0] = b0; dB[1] = b1; dB[2] = b2; dB[3] = b3;
    }
    for (int c = 0; c < nChunks - 1; c++) {
        int cur = c & 1;
        __syncthreads();
        int kb = (c + 1) << 5;
        const uint4* pa = (const uint4*)(Abase + kb);
        uint4 a0 = pa[0], a1 = pa[1];
        const uint4* pb = (const uint4*)(Bbase + kb);
        uint4 b0 = pb[0], b1 = pb[1], b2 = pb[2], b3 = pb[3];
        Frags f = frag_load_all(lA[cur], lB[cur], wrow, wcol, lane);
        uint4* dA = (uint4*)(cur ? laS[0] : laS[1]); dA[0] = a0; dA[1] = a1;
        uint4* dB = (uint4*)(cur ? lbS[0] : lbS[1]); dB[0] = b0; dB[1] = b1; dB[2] = b2; dB[3] = b3;
        wmma_all(f, acc);
    }
    {
        __syncthreads();
        int cur = (nChunks - 1) & 1;
        Frags f = frag_load_all(lA[cur], lB[cur], wrow, wcol, lane);
        wmma_all(f, acc);
    }
#endif

    // C/D layout: v8f element v at lane l -> M = (l<16 ? v : 8+v), N = l&15 (per 16x16 subtile)
    int l15 = lane & 15, lhi = (lane >> 4) & 1;
#pragma unroll
    for (int i = 0; i < 2; i++) {
#pragma unroll
        for (int j = 0; j < 4; j++) {
            int colG = n0 + wcol + j * 16 + l15;
#pragma unroll
            for (int v = 0; v < 8; v++) {
                int mi = m0 + wrow + i * 16 + lhi * 8 + v;
                if (mi >= M) continue;
                float val = acc[i][j][v];
                size_t o = (size_t)mi * N + colG;
                if constexpr (EPI == EPI_BF16) {
                    outBF[o] = f2bf(val);
                } else if constexpr (EPI == EPI_SIGMOID) {
                    outBF[o] = f2bf(sigmoidf_(val));
                } else if constexpr (EPI == EPI_MULAUX) {
                    outBF[o] = f2bf(val * bf2f(auxBF[o]));
                } else if constexpr (EPI == EPI_SIGMULAUX) {
                    outBF[o] = f2bf(bf2f(auxBF[o]) * sigmoidf_(val));
                } else if constexpr (EPI == EPI_ADDX) {
                    outF[o] = val + auxF[o];
                } else if constexpr (EPI == EPI_RELUSQ) {
                    float r = fmaxf(val, 0.f);
                    outBF[o] = f2bf(r * r);
                } else if constexpr (EPI == EPI_SCATTER) {
                    int tok = list[mi];
                    outF[(size_t)tok * C_DIM + colG] += val * scaleArr[tok];
                }
            }
        }
    }
}

// ---------------------------------------------------------------------------
extern "C" void kernel_launch(void* const* d_in, const int* in_sizes, int n_in,
                              void* d_out, int out_size, void* d_ws, size_t ws_size,
                              hipStream_t stream) {
    (void)in_sizes; (void)n_in; (void)out_size; (void)ws_size;
    const int C = C_DIM, H = H_DIM, N = N_TOK;

    const float* x          = (const float*)d_in[0];
    const float* capital    = (const float*)d_in[2];
    const float* ln1w       = (const float*)d_in[3];
    const float* ln1b       = (const float*)d_in[4];
    const float* ln2w       = (const float*)d_in[5];
    const float* ln2b       = (const float*)d_in[6];
    const float* Wr         = (const float*)d_in[7];
    const float* Wv         = (const float*)d_in[8];
    const float* Wo         = (const float*)d_in[9];
    const float* Ws         = (const float*)d_in[10];
    const float* Krw        = (const float*)d_in[11];
    const float* Vrw        = (const float*)d_in[12];
    const float* conf_rwkv  = (const float*)d_in[13];
    const float* W1         = (const float*)d_in[14];
    const float* W2         = (const float*)d_in[15];
    const float* W3         = (const float*)d_in[16];
    const float* conf_trans = (const float*)d_in[17];
    const float* w_diff     = (const float*)d_in[18];
    const float* W_aff      = (const float*)d_in[19];
    float* out = (float*)d_out;

    char* wsB = (char*)d_ws;
    size_t off = 0;
    auto alloc = [&](size_t bytes) -> void* {
        off = (off + 255) & ~(size_t)255;
        void* p = wsB + off; off += bytes; return p;
    };
    // all weight buffers hold the TRANSPOSED bf16 weights (N x K)
    unsigned short* wWr = (unsigned short*)alloc((size_t)C * C * 2);
    unsigned short* wWv = (unsigned short*)alloc((size_t)C * C * 2);
    unsigned short* wWo = (unsigned short*)alloc((size_t)C * C * 2);
    unsigned short* wWs = (unsigned short*)alloc((size_t)C * C * 2);
    unsigned short* wW1 = (unsigned short*)alloc((size_t)C * C * 2);
    unsigned short* wW2 = (unsigned short*)alloc((size_t)C * C * 2);
    unsigned short* wW3 = (unsigned short*)alloc((size_t)C * C * 2);
    unsigned short* wK0 = (unsigned short*)alloc((size_t)C * H * 2);
    unsigned short* wK1 = (unsigned short*)alloc((size_t)C * H * 2);
    unsigned short* wV0 = (unsigned short*)alloc((size_t)H * C * 2);
    unsigned short* wV1 = (unsigned short*)alloc((size_t)H * C * 2);
    unsigned short* xln    = (unsigned short*)alloc((size_t)N * C * 2);  // later reused as tg
    unsigned short* gate   = (unsigned short*)alloc((size_t)N * C * 2);  // later reused as t1
    unsigned short* gbuf   = (unsigned short*)alloc((size_t)N * C * 2);
    unsigned short* stateB = (unsigned short*)alloc((size_t)N * C * 2);
    unsigned short* hB     = (unsigned short*)alloc((size_t)N * C * 2);
    unsigned short* hrBuf  = (unsigned short*)alloc((size_t)N * H * 2);  // reused by both RWKV experts
    float* scaleA = (float*)alloc((size_t)N * 4);
    int*   lists  = (int*)alloc((size_t)3 * N * 4);
    int*   counts = (int*)alloc(256);

    auto cvtT = [&](const float* s, unsigned short* d2, int Kd, int Nd) {
        k_cvt_t<<<dim3(Kd / 32, Nd / 32), dim3(256), 0, stream>>>(s, d2, Kd, Nd);
    };
    cvtT(Wr, wWr, C, C); cvtT(Wv, wWv, C, C);
    cvtT(Wo, wWo, C, C); cvtT(Ws, wWs, C, C);
    cvtT(W1, wW1, C, C); cvtT(W2, wW2, C, C); cvtT(W3, wW3, C, C);
    cvtT(Krw,                 wK0, C, H);
    cvtT(Krw + (size_t)C * H, wK1, C, H);
    cvtT(Vrw,                 wV0, H, C);
    cvtT(Vrw + (size_t)H * C, wV1, H, C);

    // LN1
    k_layernorm<<<dim3(N), dim3(256), 0, stream>>>(x, ln1w, ln1b, xln);

    dim3 blk(128);
    dim3 gCC(N / 64, C / 128);   // (128, 8)
    dim3 gCH(N / 64, H / 128);   // (128, 32)

    // gate = sigmoid(xln @ Wr)
    k_gemm<EPI_SIGMOID, false><<<gCC, blk, 0, stream>>>(xln, wWr, C, C, nullptr, N, nullptr, nullptr, nullptr, nullptr, gate, nullptr);
    // g = gate * (xln @ Wv)
    k_gemm<EPI_MULAUX, false><<<gCC, blk, 0, stream>>>(xln, wWv, C, C, nullptr, N, nullptr, gate, nullptr, nullptr, gbuf, nullptr);
    // state = xln @ Ws
    k_gemm<EPI_BF16, false><<<gCC, blk, 0, stream>>>(xln, wWs, C, C, nullptr, N, nullptr, nullptr, nullptr, nullptr, stateB, nullptr);
    // d_out = x + g @ Wo   (x2, residual base for the expert scatter)
    k_gemm<EPI_ADDX, false><<<gCC, blk, 0, stream>>>(gbuf, wWo, C, C, nullptr, N, nullptr, nullptr, x, nullptr, nullptr, out);
    // h = LN2(x2)
    k_layernorm<<<dim3(N), dim3(256), 0, stream>>>(out, ln2w, ln2b, hB);

    // routing
    k_zero_counts<<<dim3(1), dim3(32), 0, stream>>>(counts);
    k_route<<<dim3(N), dim3(256), 0, stream>>>(hB, conf_rwkv, conf_trans, w_diff, W_aff, capital,
                                               scaleA, lists, counts);

    // RWKV experts 0,1 (token-gathered): hr = relu(h@K)^2 ; out += (hr @ V) * scale
    for (int e = 0; e < 2; e++) {
        const int* cnt = counts + e;
        const int* lst = lists + (size_t)e * N;
        unsigned short* wK = e ? wK1 : wK0;
        unsigned short* wV = e ? wV1 : wV0;
        k_gemm<EPI_RELUSQ, true ><<<gCH, blk, 0, stream>>>(hB, wK, C, H, cnt, N, lst, nullptr, nullptr, nullptr, hrBuf, nullptr);
        k_gemm<EPI_SCATTER, false><<<gCC, blk, 0, stream>>>(hrBuf, wV, H, C, cnt, N, lst, nullptr, nullptr, scaleA, nullptr, out);
    }

    // transformer expert (token-gathered): out += ((h@W1) * sigmoid(state@W2)) @ W3 * scale
    {
        const int* cnt = counts + 2;
        const int* lst = lists + (size_t)2 * N;
        unsigned short* t1 = gate;  // gate no longer needed
        unsigned short* tg = xln;   // xln no longer needed
        k_gemm<EPI_BF16, true     ><<<gCC, blk, 0, stream>>>(hB, wW1, C, C, cnt, N, lst, nullptr, nullptr, nullptr, t1, nullptr);
        k_gemm<EPI_SIGMULAUX, true><<<gCC, blk, 0, stream>>>(stateB, wW2, C, C, cnt, N, lst, t1, nullptr, nullptr, tg, nullptr);
        k_gemm<EPI_SCATTER, false ><<<gCC, blk, 0, stream>>>(tg, wW3, C, C, cnt, N, lst, nullptr, nullptr, scaleA, nullptr, out);
    }
}
